// SelfAttention_31327491457188
// MI455X (gfx1250) — matrix operand order repaired and checked by reading, not compiled, not run on hardware
//
#include <hip/hip_runtime.h>

typedef __attribute__((ext_vector_type(4)))  _Float16 v4h;
typedef __attribute__((ext_vector_type(8)))  _Float16 v8h;
typedef __attribute__((ext_vector_type(16))) _Float16 v16h;
typedef __attribute__((ext_vector_type(8)))  float    v8f;
typedef __attribute__((ext_vector_type(4)))  unsigned u32x4;
typedef __attribute__((ext_vector_type(4)))  int      i32x4;
typedef __attribute__((ext_vector_type(8)))  int      i32x8;

#define D_MODEL 1024
#define NHEAD   16
#define HD      64
#define SEQ     2048
#define BATCH   2
#define NTOK    (BATCH * SEQ)   // 4096

#if __has_builtin(__builtin_amdgcn_global_load_async_to_lds_b128)
#define HAVE_ASYNC 1
#endif
#if __has_builtin(__builtin_amdgcn_tensor_load_to_lds)
#define HAVE_TDM 1
#endif

// ---------------------------------------------------------------------------
// gfx1250 data-movement helpers (guarded; fall back to plain copies)
// ---------------------------------------------------------------------------
__device__ __forceinline__ void async_copy_b128(const _Float16* g, _Float16* l) {
#ifdef HAVE_ASYNC
  // GLOBAL_LOAD_ASYNC_TO_LDS_B128: per-lane 16B global -> LDS, tracked by ASYNCcnt
  __builtin_amdgcn_global_load_async_to_lds_b128(
      (__attribute__((address_space(1))) i32x4*)(unsigned long long)(size_t)g,
      (__attribute__((address_space(3))) i32x4*)(unsigned)(size_t)l,
      0, 0);
#else
  *(v8h*)l = *(const v8h*)g;
#endif
}

__device__ __forceinline__ void wait_async0() {
#if __has_builtin(__builtin_amdgcn_s_wait_asynccnt)
  __builtin_amdgcn_s_wait_asynccnt(0);
#else
  asm volatile("s_wait_asynccnt 0x0" ::: "memory");
#endif
}

__device__ __forceinline__ void wait_tensor0() {
#if __has_builtin(__builtin_amdgcn_s_wait_tensorcnt)
  __builtin_amdgcn_s_wait_tensorcnt(0);
#else
  asm volatile("s_wait_tensorcnt 0x0" ::: "memory");
#endif
}

#ifdef HAVE_TDM
// TDM: move a tile_w x tile_h f16 tile (row stride = row_stride elements) from
// global into LDS via one Tensor-Data-Mover descriptor (D# per ISA 08_async_tensor).
__device__ __forceinline__ void tdm_load_tile_f16(const _Float16* gsrc,
                                                  _Float16* lds,
                                                  unsigned tile_w, unsigned tile_h,
                                                  unsigned row_stride,
                                                  unsigned tensor_w, unsigned tensor_h) {
  const unsigned long long ga = (unsigned long long)(size_t)gsrc;
  const unsigned la = (unsigned)(size_t)lds;   // low 32 bits = LDS byte offset
  u32x4 g0;
  g0[0] = 1u;                                   // count=1, user descriptor
  g0[1] = la;                                   // lds_addr
  g0[2] = (unsigned)(ga & 0xFFFFFFFFu);         // global_addr[31:0]
  g0[3] = (unsigned)((ga >> 32) & 0x01FFFFFFu)  // global_addr[56:32]
          | (2u << 30);                         // type=2 ("image")
  i32x8 g1;
  g1[0] = (int)(1u << 16);                      // wg_mask=0, data_size=1 (2 bytes)
  g1[1] = (int)(tensor_w << 16);                // tensor_dim0[15:0]
  g1[2] = (int)((tensor_w >> 16) & 0xFFFFu) | (int)(tensor_h << 16);
  g1[3] = (int)((tensor_h >> 16) & 0xFFFFu) | (int)(tile_w << 16);   // tile_dim0
  g1[4] = (int)(tile_h & 0xFFFFu);              // tile_dim1 (tile_dim2=0)
  g1[5] = (int)row_stride;                      // tensor_dim0_stride[31:0]
  g1[6] = 0;                                    // stride hi / dim1_stride lo
  g1[7] = 0;
  const i32x4 z4 = {0, 0, 0, 0};
#if defined(__clang_major__) && (__clang_major__ >= 23)
  const i32x8 z8 = {0, 0, 0, 0, 0, 0, 0, 0};
  __builtin_amdgcn_tensor_load_to_lds(g0, g1, z4, z4, z8, 0);
#else
  __builtin_amdgcn_tensor_load_to_lds(g0, g1, z4, z4, 0);
#endif
}
#endif

// ---------------------------------------------------------------------------
// WMMA fragment helpers (wave32, v_wmma_f32_16x16x32_f16 layouts per CDNA5 ISA)
// ---------------------------------------------------------------------------
__device__ __forceinline__ v16h mk16(v8h lo, v8h hi) {
  return __builtin_shufflevector(lo, hi, 0,1,2,3,4,5,6,7,8,9,10,11,12,13,14,15);
}

// A fragment: 16x32 f16 tile, src = &tile[m0][k0], row-major stride lda (halfs).
__device__ __forceinline__ v16h frag_a(const _Float16* src, int lda, int lane) {
  const int hf = lane >> 4;
  const _Float16* p = src + (lane & 15) * lda + hf * 8;
  return mk16(*(const v8h*)p, *(const v8h*)(p + 16));
}

// B fragment: 32x16 f16 tile stored n-major (Bt[n][k]); src=&Bt[n0][k0].
__device__ __forceinline__ v16h frag_b(const _Float16* src, int ldb, int lane) {
  const int hf = lane >> 4;
  const _Float16* p = src + (lane & 15) * ldb + hf * 8;
  return mk16(*(const v8h*)p, *(const v8h*)(p + 16));
}

__device__ __forceinline__ v8f wmma_f16(v16h a, v16h b, v8f c) {
  return __builtin_amdgcn_wmma_f32_16x16x32_f16(false, a, false, b, (short)0, c,
                                                false, false);
}

// ---------------------------------------------------------------------------
// Kernel 1: LayerNorm (f32 in -> f16 normalized out)
// ---------------------------------------------------------------------------
__global__ void ln_kernel(const float* __restrict__ tokens,
                          const float* __restrict__ gamma,
                          const float* __restrict__ beta,
                          _Float16* __restrict__ xhat) {
  const int row = blockIdx.x;
  const int t = threadIdx.x;                       // 256 threads, 4 elems each
  const float* x = tokens + (size_t)row * D_MODEL;
  float4 v = *(const float4*)(x + t * 4);
  __shared__ float rs[256], rq[256];
  rs[t] = v.x + v.y + v.z + v.w;
  rq[t] = v.x * v.x + v.y * v.y + v.z * v.z + v.w * v.w;
  __syncthreads();
  for (int o = 128; o > 0; o >>= 1) {
    if (t < o) { rs[t] += rs[t + o]; rq[t] += rq[t + o]; }
    __syncthreads();
  }
  const float mu   = rs[0] * (1.0f / D_MODEL);
  const float var  = rq[0] * (1.0f / D_MODEL) - mu * mu;
  const float rstd = rsqrtf(var + 1e-5f);
  float4 g = *(const float4*)(gamma + t * 4);
  float4 b = *(const float4*)(beta  + t * 4);
  v4h o;
  o[0] = (_Float16)((v.x - mu) * rstd * g.x + b.x);
  o[1] = (_Float16)((v.y - mu) * rstd * g.y + b.y);
  o[2] = (_Float16)((v.z - mu) * rstd * g.z + b.z);
  o[3] = (_Float16)((v.w - mu) * rstd * g.w + b.w);
  *(v4h*)(xhat + (size_t)row * D_MODEL + t * 4) = o;
}

// ---------------------------------------------------------------------------
// Tiled GEMM core: C[128x128 block] = A(f16,[M][1024]) x W(f32,[1024][N])
// 256 threads = 8 waves; wave tile 32x64 = 2x4 wmma fragments.
// A tile staged by the Tensor Data Mover when available.
// ---------------------------------------------------------------------------
template <int N>
__device__ __forceinline__ void gemm_core(const _Float16* __restrict__ A,
                                          const float* __restrict__ W,
                                          _Float16* As,   // [128][32] f16
                                          _Float16* Bt,   // [128 n][32 k] f16
                                          v8f acc[2][4]) {
  const int t = threadIdx.x;
  const int lane = t & 31;
  const int wave = t >> 5;
  const int waveM = wave >> 1;   // 0..3
  const int waveN = wave & 1;    // 0..1
  const int bm = blockIdx.y * 128;
  const int bn = blockIdx.x * 128;
  const int arow = t >> 1, acol = (t & 1) * 16;   // A staging (fallback path)
  const int bk = t & 31, bn0 = (t >> 5) * 16;     // B staging: 16 floats/thread
  (void)arow; (void)acol;

  for (int kt = 0; kt < 1024; kt += 32) {
#ifdef HAVE_TDM
    // One TDM descriptor moves the whole 128x32 f16 A tile into LDS.
    if (wave == 0)
      tdm_load_tile_f16(A + (size_t)bm * 1024 + kt, As,
                        /*tile_w=*/32, /*tile_h=*/128,
                        /*row_stride=*/1024,
                        /*tensor_w=*/1024, /*tensor_h=*/NTOK);
#else
    {
      const _Float16* src = A + (size_t)(bm + arow) * 1024 + kt + acol;
      *(v8h*)(As + arow * 32 + acol)     = *(const v8h*)src;
      *(v8h*)(As + arow * 32 + acol + 8) = *(const v8h*)(src + 8);
    }
#endif
    // Stage W tile transposed + converted f32->f16 (Bt[n][k])
    {
      const float* src = W + (size_t)(kt + bk) * N + bn + bn0;
      #pragma unroll
      for (int j = 0; j < 16; ++j)
        Bt[(bn0 + j) * 32 + bk] = (_Float16)src[j];
    }
#ifdef HAVE_TDM
    wait_tensor0();
#endif
    __syncthreads();
    v16h aF[2], bF[4];
    #pragma unroll
    for (int i = 0; i < 2; ++i)
      aF[i] = frag_a(As + (waveM * 32 + i * 16) * 32, 32, lane);
    #pragma unroll
    for (int j = 0; j < 4; ++j)
      bF[j] = frag_b(Bt + (waveN * 64 + j * 16) * 32, 32, lane);
    #pragma unroll
    for (int i = 0; i < 2; ++i)
      #pragma unroll
      for (int j = 0; j < 4; ++j)
        acc[i][j] = wmma_f16(aF[i], bF[j], acc[i][j]);
    __syncthreads();
  }
}

// ---------------------------------------------------------------------------
// Kernel 2: QKV projection. M=4096, K=1024, N=3072. Scatters q/k/v f16.
// ---------------------------------------------------------------------------
__global__ void qkv_kernel(const _Float16* __restrict__ xhat,
                           const float* __restrict__ wqkv,
                           _Float16* __restrict__ qb,
                           _Float16* __restrict__ kb,
                           _Float16* __restrict__ vb) {
  __shared__ __align__(16) _Float16 As[128 * 32];
  __shared__ __align__(16) _Float16 Bt[128 * 32];
  v8f acc[2][4];
  #pragma unroll
  for (int i = 0; i < 2; ++i)
    #pragma unroll
    for (int j = 0; j < 4; ++j)
      acc[i][j] = v8f{0.f, 0.f, 0.f, 0.f, 0.f, 0.f, 0.f, 0.f};

  gemm_core<3 * D_MODEL>(xhat, wqkv, As, Bt, acc);

  const int lane = threadIdx.x & 31;
  const int wave = threadIdx.x >> 5;
  const int hf = lane >> 4, ln = lane & 15;
  #pragma unroll
  for (int i = 0; i < 2; ++i)
    #pragma unroll
    for (int j = 0; j < 4; ++j)
      #pragma unroll
      for (int r = 0; r < 8; ++r) {
        const int m = blockIdx.y * 128 + (wave >> 1) * 32 + i * 16 + r + 8 * hf;
        const int n = blockIdx.x * 128 + (wave & 1) * 64 + j * 16 + ln;
        const int b = m >> 11, s = m & (SEQ - 1);
        const int sel = n >> 10, h = (n >> 6) & (NHEAD - 1), hd = n & (HD - 1);
        _Float16* dst = (sel == 0) ? qb : (sel == 1) ? kb : vb;
        dst[((size_t)(b * NHEAD + h) * SEQ + s) * HD + hd] = (_Float16)acc[i][j][r];
      }
}

// ---------------------------------------------------------------------------
// Kernel 3: Flash attention. grid = (S/64, B*H), block = 128 threads (4 waves).
// K chunk staged once into LDS via async global->LDS copies (ASYNCcnt).
// ---------------------------------------------------------------------------
__global__ void attn_kernel(const _Float16* __restrict__ qg,
                            const _Float16* __restrict__ kg,
                            const _Float16* __restrict__ vg,
                            _Float16* __restrict__ attnout) {
  __shared__ __align__(16) _Float16 Kraw[64 * 64];      // K[key][hd] (B-frag ready)
  __shared__ __align__(16) _Float16 Vt[64 * 64];        // Vt[hd][key]
  __shared__ __align__(16) _Float16 Pt[4][16 * 64];     // per-wave P tile
  const int t = threadIdx.x;
  const int lane = t & 31, wave = t >> 5;
  const int hf = lane >> 4, ln = lane & 15;
  const int bh = blockIdx.y;
  const size_t base = (size_t)bh * SEQ * HD;
  const _Float16* Q = qg + base;
  const _Float16* K = kg + base;
  const _Float16* V = vg + base;
  const int q0 = blockIdx.x * 64 + wave * 16;

  // Q fragments straight from global (layout already matches A-fragment)
  v16h qA[2];
  #pragma unroll
  for (int ks = 0; ks < 2; ++ks)
    qA[ks] = frag_a(Q + (size_t)q0 * HD + ks * 32, HD, lane);

  v8f o[4];
  #pragma unroll
  for (int j = 0; j < 4; ++j) o[j] = v8f{0.f, 0.f, 0.f, 0.f, 0.f, 0.f, 0.f, 0.f};
  float mrow[8], lrow[8];
  #pragma unroll
  for (int r = 0; r < 8; ++r) { mrow[r] = -3.0e38f; lrow[r] = 0.f; }

  const float scale = 0.125f;                 // HD^-0.5
  const int vrow = t >> 1, vcol = (t & 1) * 32;

  for (int c = 0; c < SEQ; c += 64) {
    // Async-stage K chunk raw (row-major == B-fragment layout): 64B/thread
    {
      const _Float16* src = K + (size_t)(c + vrow) * HD + vcol;
      _Float16* dst = Kraw + vrow * 64 + vcol;
      #pragma unroll
      for (int u = 0; u < 4; ++u)
        async_copy_b128(src + u * 8, dst + u * 8);
    }
    // Stage V chunk transposed: Vt[hd][key]
    {
      const _Float16* src = V + (size_t)(c + vrow) * HD + vcol;
      v8h tmp[4];
      #pragma unroll
      for (int u = 0; u < 4; ++u) tmp[u] = *(const v8h*)(src + u * 8);
      #pragma unroll
      for (int j = 0; j < 32; ++j)
        Vt[(vcol + j) * 64 + vrow] = tmp[j >> 3][j & 7];
    }
    wait_async0();
    __syncthreads();

    // Scores: S = Q x K^T
    v8f s0[4];
    #pragma unroll
    for (int j = 0; j < 4; ++j) s0[j] = v8f{0.f, 0.f, 0.f, 0.f, 0.f, 0.f, 0.f, 0.f};
    #pragma unroll
    for (int ks = 0; ks < 2; ++ks)
      #pragma unroll
      for (int j = 0; j < 4; ++j) {
        v16h kB = frag_b(Kraw + (j * 16) * 64 + ks * 32, 64, lane);
        s0[j] = wmma_f16(qA[ks], kB, s0[j]);
      }

    // Online softmax, per row (rows live across 16-lane half-waves)
    #pragma unroll
    for (int r = 0; r < 8; ++r) {
      float cm = -3.0e38f;
      #pragma unroll
      for (int j = 0; j < 4; ++j) {
        s0[j][r] *= scale;
        cm = fmaxf(cm, s0[j][r]);
      }
      #pragma unroll
      for (int msk = 8; msk >= 1; msk >>= 1)
        cm = fmaxf(cm, __shfl_xor(cm, msk, 16));
      const float mnew = fmaxf(mrow[r], cm);
      const float corr = __expf(mrow[r] - mnew);
      float rsum = 0.f;
      #pragma unroll
      for (int j = 0; j < 4; ++j) {
        const float p = __expf(s0[j][r] - mnew);
        s0[j][r] = p;
        rsum += p;
      }
      #pragma unroll
      for (int msk = 8; msk >= 1; msk >>= 1)
        rsum += __shfl_xor(rsum, msk, 16);
      lrow[r] = lrow[r] * corr + rsum;
      mrow[r] = mnew;
      #pragma unroll
      for (int j = 0; j < 4; ++j) o[j][r] *= corr;
    }

    // Re-layout P (C-layout f32) -> A-fragment f16 via per-wave LDS tile
    #pragma unroll
    for (int j = 0; j < 4; ++j)
      #pragma unroll
      for (int r = 0; r < 8; ++r)
        Pt[wave][(r + 8 * hf) * 64 + j * 16 + ln] = (_Float16)s0[j][r];

    // O += P x V
    #pragma unroll
    for (int ks = 0; ks < 2; ++ks) {
      v16h pA = frag_a(&Pt[wave][0] + ks * 32, 64, lane);
      #pragma unroll
      for (int j = 0; j < 4; ++j) {
        v16h vB = frag_b(Vt + (j * 16) * 64 + ks * 32, 64, lane);
        o[j] = wmma_f16(pA, vB, o[j]);
      }
    }
    __syncthreads();
  }

  // Normalize and write to attnbuf[token][h*64+hd] (f16, feeds proj GEMM)
  const int b = bh >> 4, h = bh & (NHEAD - 1);
  #pragma unroll
  for (int j = 0; j < 4; ++j)
    #pragma unroll
    for (int r = 0; r < 8; ++r) {
      const int s = q0 + r + 8 * hf;
      const int col = h * HD + j * 16 + ln;
      attnout[(size_t)(b * SEQ + s) * D_MODEL + col] = (_Float16)(o[j][r] / lrow[r]);
    }
}

// ---------------------------------------------------------------------------
// Kernel 4: Output projection + bias + residual. M=4096, K=1024, N=1024, f32 out
// ---------------------------------------------------------------------------
__global__ void proj_kernel(const _Float16* __restrict__ attn,
                            const float* __restrict__ wproj,
                            const float* __restrict__ bproj,
                            const float* __restrict__ tokens,
                            float* __restrict__ out) {
  __shared__ __align__(16) _Float16 As[128 * 32];
  __shared__ __align__(16) _Float16 Bt[128 * 32];
  v8f acc[2][4];
  #pragma unroll
  for (int i = 0; i < 2; ++i)
    #pragma unroll
    for (int j = 0; j < 4; ++j)
      acc[i][j] = v8f{0.f, 0.f, 0.f, 0.f, 0.f, 0.f, 0.f, 0.f};

  gemm_core<D_MODEL>(attn, wproj, As, Bt, acc);

  const int lane = threadIdx.x & 31;
  const int wave = threadIdx.x >> 5;
  const int hf = lane >> 4, ln = lane & 15;
  #pragma unroll
  for (int i = 0; i < 2; ++i)
    #pragma unroll
    for (int j = 0; j < 4; ++j)
      #pragma unroll
      for (int r = 0; r < 8; ++r) {
        const int m = blockIdx.y * 128 + (wave >> 1) * 32 + i * 16 + r + 8 * hf;
        const int n = blockIdx.x * 128 + (wave & 1) * 64 + j * 16 + ln;
        const size_t idx = (size_t)m * D_MODEL + n;
        out[idx] = acc[i][j][r] + bproj[n] + tokens[idx];
      }
}

// ---------------------------------------------------------------------------
extern "C" void kernel_launch(void* const* d_in, const int* in_sizes, int n_in,
                              void* d_out, int out_size, void* d_ws, size_t ws_size,
                              hipStream_t stream) {
  (void)in_sizes; (void)n_in; (void)out_size; (void)ws_size;
  const float* tokens = (const float*)d_in[0];
  const float* gamma  = (const float*)d_in[1];
  const float* beta   = (const float*)d_in[2];
  const float* wqkv   = (const float*)d_in[3];
  const float* wproj  = (const float*)d_in[4];
  const float* bproj  = (const float*)d_in[5];
  float* out = (float*)d_out;

  char* ws = (char*)d_ws;
  const size_t SZ = (size_t)NTOK * D_MODEL * sizeof(_Float16);  // 8 MB each
  _Float16* xhat = (_Float16*)(ws);
  _Float16* qb   = (_Float16*)(ws + 1 * SZ);
  _Float16* kb   = (_Float16*)(ws + 2 * SZ);
  _Float16* vb   = (_Float16*)(ws + 3 * SZ);
  _Float16* ab   = (_Float16*)(ws + 4 * SZ);

  ln_kernel<<<NTOK, 256, 0, stream>>>(tokens, gamma, beta, xhat);
  qkv_kernel<<<dim3(3 * D_MODEL / 128, NTOK / 128), 256, 0, stream>>>(
      xhat, wqkv, qb, kb, vb);
  attn_kernel<<<dim3(SEQ / 64, BATCH * NHEAD), 128, 0, stream>>>(qb, kb, vb, ab);
  proj_kernel<<<dim3(D_MODEL / 128, NTOK / 128), 256, 0, stream>>>(
      ab, wproj, bproj, tokens, out);
}
